// MoE_26061861552327
// MI455X (gfx1250) — compile-verified, matrix operand order checked by reading, not run on hardware
//
#include <hip/hip_runtime.h>
#include <hip/hip_bf16.h>
#include <math.h>

// ---- CDNA5 WMMA types ----
typedef __attribute__((ext_vector_type(16))) __bf16 v16bf;
typedef __attribute__((ext_vector_type(8)))  __bf16 v8bf;
typedef __attribute__((ext_vector_type(8)))  float  v8f;

// Problem constants (match setup_inputs: B=1, S=2048, H=3072, E=4, I=8192, top_k=2)
constexpr int kN = 2048;
constexpr int kH = 3072;
constexpr int kE = 4;
constexpr int kI = 8192;
constexpr int kTopK = 2;

__device__ __forceinline__ v8f wmma_bf16(v16bf a, v16bf b, v8f c) {
  // v_wmma_f32_16x16x32_bf16  D = A*B + C
  return __builtin_amdgcn_wmma_f32_16x16x32_bf16(false, a, false, b, (short)0, c,
                                                 false, false);
}

// Async DMA: global -> LDS, 16 bytes per lane, tracked with ASYNCcnt.
// LDS destination address comes from vdst; generic LDS pointer low 32 bits = LDS offset.
__device__ __forceinline__ void async_copy_b128(const void* gsrc, void* lds_dst) {
  unsigned ldsa = (unsigned)(uintptr_t)lds_dst;
  asm volatile("global_load_async_to_lds_b128 %0, %1, off"
               :
               : "v"(ldsa), "v"(gsrc)
               : "memory");
}
__device__ __forceinline__ void wait_async0() {
  asm volatile("s_wait_asynccnt 0x0" ::: "memory");
}

__device__ __forceinline__ float fast_sigmoid(float x) {
#if __has_builtin(__builtin_amdgcn_rcpf)
  return __builtin_amdgcn_rcpf(1.0f + __expf(-x));
#else
  return 1.0f / (1.0f + __expf(-x));
#endif
}

// Build a B fragment (32x16, K x N) for one lane from an fp32 weight row.
// Lane L holds column n = L&15 (weight row), lane group g = L>>4.
// Element j<8  -> K = 8*g + j      (8 contiguous floats)
// Element j>=8 -> K = 16 + 8*g + (j-8)
__device__ __forceinline__ v16bf load_b_frag_f32(const float* __restrict__ p, int g) {
  const float4 f0 = *(const float4*)(p + 8 * g + 0);
  const float4 f1 = *(const float4*)(p + 8 * g + 4);
  const float4 f2 = *(const float4*)(p + 8 * g + 16);
  const float4 f3 = *(const float4*)(p + 8 * g + 20);
  v16bf b;
  b[0]  = (__bf16)f0.x; b[1]  = (__bf16)f0.y; b[2]  = (__bf16)f0.z; b[3]  = (__bf16)f0.w;
  b[4]  = (__bf16)f1.x; b[5]  = (__bf16)f1.y; b[6]  = (__bf16)f1.z; b[7]  = (__bf16)f1.w;
  b[8]  = (__bf16)f2.x; b[9]  = (__bf16)f2.y; b[10] = (__bf16)f2.z; b[11] = (__bf16)f2.w;
  b[12] = (__bf16)f3.x; b[13] = (__bf16)f3.y; b[14] = (__bf16)f3.z; b[15] = (__bf16)f3.w;
  return b;
}

// Build an A fragment (16x32, M x K) from a 16-row x 32-K bf16 LDS tile.
__device__ __forceinline__ v16bf load_a_frag_lds(const __bf16* sA, int m, int g) {
  v8bf lo = *(const v8bf*)(sA + m * 32 + 8 * g);
  v8bf hi = *(const v8bf*)(sA + m * 32 + 16 + 8 * g);
  return __builtin_shufflevector(lo, hi, 0, 1, 2, 3, 4, 5, 6, 7,
                                 8, 9, 10, 11, 12, 13, 14, 15);
}

// ---------------------------------------------------------------------------
// 1) Gating: logits -> softmax -> top-2; atomic-append token into per-expert
//    compacted lists together with its gate probability.
// ---------------------------------------------------------------------------
__global__ __launch_bounds__(128) void gate_topk_kernel(
    const float* __restrict__ x, const float* __restrict__ gate_w,
    int* __restrict__ cnt, int* __restrict__ ltok, float* __restrict__ lval) {
  __shared__ float red[kE][128];
  const int n = blockIdx.x;
  const int tid = threadIdx.x;
  const float* xr = x + (size_t)n * kH;
  float p0 = 0.f, p1 = 0.f, p2 = 0.f, p3 = 0.f;
  for (int h = tid; h < kH; h += 128) {
    float xv = xr[h];
    p0 += xv * gate_w[0 * kH + h];
    p1 += xv * gate_w[1 * kH + h];
    p2 += xv * gate_w[2 * kH + h];
    p3 += xv * gate_w[3 * kH + h];
  }
  red[0][tid] = p0; red[1][tid] = p1; red[2][tid] = p2; red[3][tid] = p3;
  __syncthreads();
  for (int s = 64; s > 0; s >>= 1) {
    if (tid < s) {
      for (int e = 0; e < kE; ++e) red[e][tid] += red[e][tid + s];
    }
    __syncthreads();
  }
  if (tid == 0) {
    float l[kE];
    for (int e = 0; e < kE; ++e) l[e] = red[e][0];
    int i1 = 0;
    for (int e = 1; e < kE; ++e) if (l[e] > l[i1]) i1 = e;
    int i2 = -1;
    for (int e = 0; e < kE; ++e)
      if (e != i1 && (i2 < 0 || l[e] > l[i2])) i2 = e;
    float m = l[i1], sum = 0.f, pr[kE];
    for (int e = 0; e < kE; ++e) { pr[e] = __expf(l[e] - m); sum += pr[e]; }
    float inv = 1.f / sum;
    int sel[2] = {i1, i2};
    for (int k = 0; k < kTopK; ++k) {
      int e = sel[k];
      int pos = atomicAdd(&cnt[e], 1);
      ltok[e * kN + pos] = n;
      lval[e * kN + pos] = pr[e] * inv;
    }
  }
}

// ---------------------------------------------------------------------------
// 2) Exclusive prefix over 4 expert counts -> compacted hidden-row offsets.
// ---------------------------------------------------------------------------
__global__ void offsets_kernel(const int* __restrict__ cnt, int* __restrict__ offs) {
  if (threadIdx.x == 0) {
    int a = 0;
    for (int e = 0; e < kE; ++e) { offs[e] = a; a += cnt[e]; }
  }
}

// ---------------------------------------------------------------------------
// 3) x fp32 -> bf16
// ---------------------------------------------------------------------------
__global__ __launch_bounds__(256) void convert_x_kernel(
    const float* __restrict__ x, __bf16* __restrict__ xb) {
  int i = blockIdx.x * 256 + threadIdx.x;
  xb[i] = (__bf16)x[i];
}

// ---------------------------------------------------------------------------
// 4) Up-projection grouped GEMM + fused SwiGLU.
//    Block: 256 threads = 8 waves. Tile: M=128 rows (8 WMMA subtiles/wave),
//    each wave owns 16 gate cols + the matching 16 up cols (same A reused).
//    A tile double-buffered in LDS, staged by async global->LDS DMA.
//    grid = (kI/128, kN/128, kE); blocks past the expert's row count exit.
// ---------------------------------------------------------------------------
__global__ __launch_bounds__(256) void moe_up_kernel(
    const float* __restrict__ w_gate_up, const __bf16* __restrict__ xb,
    const int* __restrict__ cnt, const int* __restrict__ offs,
    const int* __restrict__ ltok, __bf16* __restrict__ hidden) {
  const int e = blockIdx.z;
  const int mcount = cnt[e];
  const int m0 = blockIdx.y * 128;
  if (m0 >= mcount) return;

  __shared__ int sTok[128];
  __shared__ __attribute__((aligned(16))) __bf16 sA[2][128 * 32];

  const int tid = threadIdx.x;
  if (tid < 128) {
    int r = m0 + tid;
    sTok[tid] = ltok[e * kN + (r < mcount ? r : 0)];
  }
  __syncthreads();

  const int lane = tid & 31;
  const int wv   = tid >> 5;
  const int g    = lane >> 4;
  const int n16  = lane & 15;
  const int cg   = blockIdx.x * 128 + wv * 16;  // gate column base for this wave

  const float* wgp = w_gate_up + ((size_t)e * (2 * kI) + cg + n16) * (size_t)kH;
  const float* wup = w_gate_up + ((size_t)e * (2 * kI) + kI + cg + n16) * (size_t)kH;

  v8f accg[8], accu[8];
#pragma unroll
  for (int s = 0; s < 8; ++s) { accg[s] = (v8f)0.f; accu[s] = (v8f)0.f; }

  const int arow = tid >> 1;            // row this thread stages (0..127)
  const int acol = (tid & 1) * 16;      // K half it stages
  const __bf16* xrow = xb + (size_t)sTok[arow] * kH + acol;

  // Prologue: DMA chunk 0 into buffer 0.
  async_copy_b128(xrow + 0, &sA[0][arow * 32 + acol + 0]);
  async_copy_b128(xrow + 8, &sA[0][arow * 32 + acol + 8]);
  wait_async0();
  __syncthreads();

  int p = 0;
  for (int k0 = 0; k0 < kH; k0 += 32) {
    // Kick off DMA of the next A chunk into the other buffer (overlaps WMMAs).
    int kn = (k0 + 32 < kH) ? (k0 + 32) : k0;
    async_copy_b128(xrow + kn + 0, &sA[p ^ 1][arow * 32 + acol + 0]);
    async_copy_b128(xrow + kn + 8, &sA[p ^ 1][arow * 32 + acol + 8]);

    // Speculative prefetch of the weight stream two chunks ahead.
    __builtin_prefetch(wgp + k0 + 64, 0, 0);
    __builtin_prefetch(wup + k0 + 64, 0, 0);

    v16bf bg = load_b_frag_f32(wgp + k0, g);
    v16bf bu = load_b_frag_f32(wup + k0, g);
#pragma unroll
    for (int s = 0; s < 8; ++s) {
      v16bf a = load_a_frag_lds(sA[p], s * 16 + n16, g);
      accg[s] = wmma_bf16(a, bg, accg[s]);
      accu[s] = wmma_bf16(a, bu, accu[s]);
    }

    wait_async0();
    __syncthreads();
    p ^= 1;
  }

  // Epilogue: hidden = up * silu(gate), stored bf16 into compacted rows.
  const size_t hbase = (size_t)offs[e];
  const size_t col = (size_t)(blockIdx.x * 128 + wv * 16 + n16);
#pragma unroll
  for (int s = 0; s < 8; ++s) {
#pragma unroll
    for (int v = 0; v < 8; ++v) {
      int R = m0 + s * 16 + v + 8 * g;   // C/D layout: row = v + 8*(lane>>4)
      if (R < mcount) {
        float gv = accg[s][v];
        float uv = accu[s][v];
        float h = uv * gv * fast_sigmoid(gv);
        hidden[(hbase + (size_t)R) * (size_t)kI + col] = (__bf16)h;
      }
    }
  }
}

// ---------------------------------------------------------------------------
// 5) Down-projection grouped GEMM; epilogue scales rows by gate prob and
//    atomically accumulates into the fp32 output (2 experts hit each token).
//    grid = (kH/128, kN/128, kE).
// ---------------------------------------------------------------------------
__global__ __launch_bounds__(256) void moe_down_kernel(
    const float* __restrict__ w_down, const __bf16* __restrict__ hidden,
    const int* __restrict__ cnt, const int* __restrict__ offs,
    const int* __restrict__ ltok, const float* __restrict__ lval,
    float* __restrict__ out) {
  const int e = blockIdx.z;
  const int mcount = cnt[e];
  const int m0 = blockIdx.y * 128;
  if (m0 >= mcount) return;

  __shared__ int   sTok[128];
  __shared__ float sVal[128];
  __shared__ __attribute__((aligned(16))) __bf16 sA[2][128 * 32];

  const int tid = threadIdx.x;
  if (tid < 128) {
    int r = m0 + tid;
    int idx = e * kN + (r < mcount ? r : 0);
    sTok[tid] = ltok[idx];
    sVal[tid] = lval[idx];
  }
  __syncthreads();

  const int lane = tid & 31;
  const int wv   = tid >> 5;
  const int g    = lane >> 4;
  const int n16  = lane & 15;
  const int h0   = blockIdx.x * 128 + wv * 16;

  const float* wdp = w_down + ((size_t)e * kH + h0 + n16) * (size_t)kI;

  v8f acc[8];
#pragma unroll
  for (int s = 0; s < 8; ++s) acc[s] = (v8f)0.f;

  const int arow = tid >> 1;
  const int acol = (tid & 1) * 16;
  const int hrow = (m0 + arow < mcount) ? (m0 + arow) : (mcount - 1);
  const __bf16* arp = hidden + ((size_t)offs[e] + (size_t)hrow) * (size_t)kI + acol;

  async_copy_b128(arp + 0, &sA[0][arow * 32 + acol + 0]);
  async_copy_b128(arp + 8, &sA[0][arow * 32 + acol + 8]);
  wait_async0();
  __syncthreads();

  int p = 0;
  for (int k0 = 0; k0 < kI; k0 += 32) {
    int kn = (k0 + 32 < kI) ? (k0 + 32) : k0;
    async_copy_b128(arp + kn + 0, &sA[p ^ 1][arow * 32 + acol + 0]);
    async_copy_b128(arp + kn + 8, &sA[p ^ 1][arow * 32 + acol + 8]);

    __builtin_prefetch(wdp + k0 + 64, 0, 0);

    v16bf b = load_b_frag_f32(wdp + k0, g);
#pragma unroll
    for (int s = 0; s < 8; ++s) {
      v16bf a = load_a_frag_lds(sA[p], s * 16 + n16, g);
      acc[s] = wmma_bf16(a, b, acc[s]);
    }

    wait_async0();
    __syncthreads();
    p ^= 1;
  }

#pragma unroll
  for (int s = 0; s < 8; ++s) {
#pragma unroll
    for (int v = 0; v < 8; ++v) {
      int rloc = s * 16 + v + 8 * g;
      int R = m0 + rloc;
      if (R < mcount) {
        int tok  = sTok[rloc];
        float sc = sVal[rloc];
        atomicAdd(&out[(size_t)tok * kH + (size_t)(h0 + n16)], acc[s][v] * sc);
      }
    }
  }
}

// ---------------------------------------------------------------------------
// Launcher
// ---------------------------------------------------------------------------
extern "C" void kernel_launch(void* const* d_in, const int* in_sizes, int n_in,
                              void* d_out, int out_size, void* d_ws, size_t ws_size,
                              hipStream_t stream) {
  const float* x         = (const float*)d_in[0];  // [N, H]
  const float* gate_w    = (const float*)d_in[1];  // [E, H]
  const float* w_gate_up = (const float*)d_in[2];  // [E, 2I, H]
  const float* w_down    = (const float*)d_in[3];  // [E, H, I]
  float* out = (float*)d_out;                      // [N, H] fp32

  // Workspace carve-up (~80 MB)
  char* ws = (char*)d_ws;
  size_t o = 0;
  int*   cnt  = (int*)(ws + o);   o += 256;
  int*   offs = (int*)(ws + o);   o += 256;
  int*   ltok = (int*)(ws + o);   o += (size_t)kE * kN * sizeof(int);
  float* lval = (float*)(ws + o); o += (size_t)kE * kN * sizeof(float);
  o = (o + 255) & ~(size_t)255;
  __bf16* xb = (__bf16*)(ws + o); o += (size_t)kN * kH * sizeof(__bf16);
  o = (o + 255) & ~(size_t)255;
  __bf16* hidden = (__bf16*)(ws + o);  // [topk*N, I] compacted

  hipMemsetAsync(cnt, 0, 256, stream);
  hipMemsetAsync(out, 0, (size_t)kN * kH * sizeof(float), stream);

  gate_topk_kernel<<<kN, 128, 0, stream>>>(x, gate_w, cnt, ltok, lval);
  offsets_kernel<<<1, 32, 0, stream>>>(cnt, offs);
  convert_x_kernel<<<(kN * kH) / 256, 256, 0, stream>>>(x, xb);
  moe_up_kernel<<<dim3(kI / 128, kN / 128, kE), 256, 0, stream>>>(
      w_gate_up, xb, cnt, offs, ltok, hidden);
  moe_down_kernel<<<dim3(kH / 128, kN / 128, kE), 256, 0, stream>>>(
      w_down, hidden, cnt, offs, ltok, lval, out);
}
